// MultiModalFusion_47107201302653
// MI455X (gfx1250) — matrix operand order repaired and checked by reading, not compiled, not run on hardware
//
#include <hip/hip_runtime.h>

// ---- problem constants ----
#define B_ 16
#define C_ 256
#define S_ 4096
#define L_ 256
#define E_ 256
#define NH_ 4
#define HD_ 64
#define VSTRIDE 272   // Vt row stride in bf16 elems (16B-aligned: 544B)

typedef __bf16 bf16_t;
typedef bf16_t v16bf __attribute__((ext_vector_type(16)));
typedef bf16_t v8bf  __attribute__((ext_vector_type(8)));
typedef float v8f __attribute__((ext_vector_type(8)));

static __device__ __forceinline__ v8bf ld8(const bf16_t* p) {
  return *(const v8bf*)p;                    // 16B vector load (b128)
}
static __device__ __forceinline__ void st8(bf16_t* p, v8bf v) {
  *(v8bf*)p = v;                             // 16B vector store (b128)
}
static __device__ __forceinline__ v16bf cat8(v8bf lo, v8bf hi) {
  return __builtin_shufflevector(lo, hi, 0, 1, 2, 3, 4, 5, 6, 7,
                                 8, 9, 10, 11, 12, 13, 14, 15);
}
// A-fragment (16x32 bf16): lane=half*16+m, elems 0..15 = K half*16+0..15.
// base points at (row m, k=0).
static __device__ __forceinline__ v16bf load_afrag(const bf16_t* base, int half) {
  return cat8(ld8(base + half * 16), ld8(base + half * 16 + 8));
}
// B-fragment where the 16 elements (k = half*16+0..15 at fixed n) are contiguous.
static __device__ __forceinline__ v16bf load_bfrag(const bf16_t* p) {
  return cat8(ld8(p), ld8(p + 8));
}

// ================= LayerNorm of img_x ([B,C,S] -> qn bf16 [B,S,C]) ==========
__global__ __launch_bounds__(256) void ln_img_kernel(
    const float* __restrict__ img, const float* __restrict__ lw,
    const float* __restrict__ lb, bf16_t* __restrict__ qn) {
  extern __shared__ float ln_smem[];            // 256 * 65 floats (padded)
  const int b = blockIdx.x;
  const int s0 = blockIdx.y * 64;
  const int t = threadIdx.x;
  for (int i = 0; i < 64; ++i) {
    int idx = t + i * 256;                      // 0..16383
    int c = idx >> 6, j = idx & 63;
    ln_smem[c * 65 + j] = img[((size_t)b * C_ + c) * S_ + s0 + j];
  }
  __syncthreads();
  const int lane = t & 31, w = t >> 5;
  for (int it = 0; it < 8; ++it) {
    int j = w * 8 + it;
    float v[8];
    float sum = 0.f, sq = 0.f;
#pragma unroll
    for (int k = 0; k < 8; ++k) {
      int c = lane + k * 32;
      float x = ln_smem[c * 65 + j];
      v[k] = x; sum += x; sq += x * x;
    }
#pragma unroll
    for (int m = 1; m < 32; m <<= 1) {
      sum += __shfl_xor(sum, m, 32);
      sq  += __shfl_xor(sq, m, 32);
    }
    float mu = sum * (1.f / 256.f);
    float var = sq * (1.f / 256.f) - mu * mu;
    float rs = rsqrtf(var + 1e-5f);
#pragma unroll
    for (int k = 0; k < 8; ++k) {
      int c = lane + k * 32;
      qn[((size_t)b * S_ + s0 + j) * C_ + c] =
          (bf16_t)((v[k] - mu) * rs * lw[c] + lb[c]);
    }
  }
}

// ================= LayerNorm of refs ([B,L,256] -> kn bf16) =================
__global__ __launch_bounds__(256) void ln_txt_kernel(
    const float* __restrict__ refs, const float* __restrict__ lw,
    const float* __restrict__ lb, bf16_t* __restrict__ kn) {
  const int t = threadIdx.x, lane = t & 31, w = t >> 5;
  const int row = blockIdx.x * 8 + w;
  float v[8];
  float sum = 0.f, sq = 0.f;
#pragma unroll
  for (int k = 0; k < 8; ++k) {
    float x = refs[(size_t)row * 256 + lane + k * 32];
    v[k] = x; sum += x; sq += x * x;
  }
#pragma unroll
  for (int m = 1; m < 32; m <<= 1) {
    sum += __shfl_xor(sum, m, 32);
    sq  += __shfl_xor(sq, m, 32);
  }
  float mu = sum * (1.f / 256.f);
  float var = sq * (1.f / 256.f) - mu * mu;
  float rs = rsqrtf(var + 1e-5f);
#pragma unroll
  for (int k = 0; k < 8; ++k) {
    int c = lane + k * 32;
    kn[(size_t)row * 256 + c] = (bf16_t)((v[k] - mu) * rs * lw[c] + lb[c]);
  }
}

// ================= AdaLN modulation: silu(re) @ W_ada + b_ada ===============
__global__ __launch_bounds__(256) void mod_kernel(
    const float* __restrict__ re, const float* __restrict__ Wa,
    const float* __restrict__ ba, float* __restrict__ mod) {
  int b = blockIdx.x, j = blockIdx.y * 256 + threadIdx.x;
  float acc = ba[j];
  for (int c = 0; c < 256; ++c) {
    float x = re[b * 256 + c];
    float s = x / (1.f + __expf(-x));
    acc += s * Wa[(size_t)c * 768 + j];
  }
  mod[b * 768 + j] = acc;
}

// ========== generic bf16 WMMA GEMM: C[M,256] = A[M,256] @ W[256,256] + b ====
__global__ __launch_bounds__(256) void gemm_bf16_256(
    const bf16_t* __restrict__ A, const float* __restrict__ W,
    const float* __restrict__ bias, bf16_t* __restrict__ Cm) {
  __shared__ __align__(32) bf16_t As[128 * 32];
  __shared__ __align__(32) bf16_t Ws[32 * 64];   // frag-major
  const int t = threadIdx.x, lane = t & 31, w = t >> 5;
  const int half = lane >> 4, nn = lane & 15;
  const int m0 = blockIdx.x * 128, n0 = blockIdx.y * 64;
  const int cW = t & 63;          // W-stage: column within tile
  const int oct = t >> 6;         // W-stage: group of 8 k-rows
  const int wdst = (((cW >> 4) * 32) + ((oct >> 1) * 16) + (cW & 15)) * 16 + (oct & 1) * 8;
  v8f acc[4] = {};
  for (int kk = 0; kk < 256; kk += 32) {
    { // stage A tile [128 x 32]: two b128 loads + two b128 stores per thread
      int row = t >> 1, seg = (t & 1) * 16;
      const bf16_t* src = A + (size_t)(m0 + row) * 256 + kk + seg;
      st8(As + row * 32 + seg, ld8(src));
      st8(As + row * 32 + seg + 8, ld8(src + 8));
      if (kk + 32 < 256) __builtin_prefetch(src + 32, 0, 1);
    }
    { // stage W tile [32 x 64] -> frag-major bf16: one b128 store per thread
      const float* wsrc = W + (size_t)(kk + oct * 8) * 256 + n0 + cW;
      v8bf tmp;
#pragma unroll
      for (int q = 0; q < 8; ++q) tmp[q] = (bf16_t)wsrc[(size_t)q * 256];
      st8(Ws + wdst, tmp);
    }
    __syncthreads();
    v16bf a = load_afrag(As + (w * 16 + nn) * 32, half);
#pragma unroll
    for (int nt = 0; nt < 4; ++nt) {
      v16bf bfr = load_bfrag(Ws + (nt * 32 + lane) * 16);
      acc[nt] = __builtin_amdgcn_wmma_f32_16x16x32_bf16(
          false, a, false, bfr, (short)0, acc[nt], false, false);
    }
    __syncthreads();
  }
#pragma unroll
  for (int nt = 0; nt < 4; ++nt)
#pragma unroll
    for (int r = 0; r < 8; ++r) {
      int row = m0 + w * 16 + r + 8 * half;
      int col = n0 + nt * 16 + nn;
      Cm[(size_t)row * 256 + col] = (bf16_t)(acc[nt][r] + bias[col]);
    }
}

// ================= attention: softmax(QK^T/8 + mask) @ V ====================
__global__ __launch_bounds__(256) void attn_kernel(
    const bf16_t* __restrict__ Q, const bf16_t* __restrict__ K,
    const bf16_t* __restrict__ V, const unsigned char* __restrict__ masks,
    bf16_t* __restrict__ O) {
  extern __shared__ bf16_t attn_smem[];           // 100352 bytes dynamic
  bf16_t* Vt  = attn_smem;                        // [64][VSTRIDE] transposed V
  bf16_t* Ksh = attn_smem + 64 * VSTRIDE;         // [256 x 64]
  bf16_t* Psh = attn_smem + 64 * VSTRIDE;         // overlaps Ksh, [8][16][256]
  const int b = blockIdx.x, h = blockIdx.y, qb = blockIdx.z;
  const int t = threadIdx.x, lane = t & 31, w = t >> 5;
  const int half = lane >> 4, nn = lane & 15;
  for (int i = 0; i < 64; ++i) {                  // stage K natural, V transposed
    int idx = t + i * 256;
    int j = idx >> 6, d = idx & 63;               // coalesced along d
    bf16_t kv = K[((size_t)b * L_ + j) * E_ + h * HD_ + d];
    bf16_t vv = V[((size_t)b * L_ + j) * E_ + h * HD_ + d];
    Ksh[j * 64 + d] = kv;
    Vt[d * VSTRIDE + j] = vv;
  }
  __syncthreads();
  const int s0 = qb * 128 + w * 16;
  const size_t qrow0 = (size_t)b * S_ + s0;
  v8f sc[16] = {};
  for (int d0 = 0; d0 < HD_; d0 += 32) {          // scores = Q @ K^T
    v16bf aq = load_afrag(Q + (qrow0 + nn) * E_ + h * HD_ + d0, half);
#pragma unroll
    for (int tt = 0; tt < 16; ++tt) {
      v16bf bk = load_bfrag(Ksh + (tt * 16 + nn) * 64 + d0 + half * 16);
      sc[tt] = __builtin_amdgcn_wmma_f32_16x16x32_bf16(
          false, aq, false, bk, (short)0, sc[tt], false, false);
    }
  }
  __syncthreads();                                 // Ksh reads done (Psh aliases)
  const float NEG = -__builtin_inff();
#pragma unroll
  for (int tt = 0; tt < 16; ++tt) {                // scale + key padding mask
    bool msk = masks[b * L_ + tt * 16 + nn] != 0;
#pragma unroll
    for (int r = 0; r < 8; ++r) {
      float x = sc[tt][r] * 0.125f;                // 1/sqrt(64)
      sc[tt][r] = msk ? NEG : x;
    }
  }
  float rinv[8];
#pragma unroll
  for (int r = 0; r < 8; ++r) {                    // row-wise softmax
    float m = NEG;
#pragma unroll
    for (int tt = 0; tt < 16; ++tt) m = fmaxf(m, sc[tt][r]);
#pragma unroll
    for (int xm = 1; xm < 16; xm <<= 1) m = fmaxf(m, __shfl_xor(m, xm, 32));
    float s = 0.f;
#pragma unroll
    for (int tt = 0; tt < 16; ++tt) {
      float e = __expf(sc[tt][r] - m);
      sc[tt][r] = e; s += e;
    }
#pragma unroll
    for (int xm = 1; xm < 16; xm <<= 1) s += __shfl_xor(s, xm, 32);
    rinv[r] = 1.f / s;
  }
#pragma unroll
  for (int tt = 0; tt < 16; ++tt)                  // P (bf16, normalized) -> LDS
#pragma unroll
    for (int r = 0; r < 8; ++r)
      Psh[w * 4096 + (r + 8 * half) * 256 + tt * 16 + nn] =
          (bf16_t)(sc[tt][r] * rinv[r]);
  __syncthreads();
  v8f oa[4] = {};
  for (int kt = 0; kt < 8; ++kt) {                 // O = P @ V
    v16bf ap = load_afrag(Psh + w * 4096 + nn * 256 + kt * 32, half);
#pragma unroll
    for (int nt = 0; nt < 4; ++nt) {
      v16bf bvf = load_bfrag(Vt + (nt * 16 + nn) * VSTRIDE + kt * 32 + half * 16);
      oa[nt] = __builtin_amdgcn_wmma_f32_16x16x32_bf16(
          false, ap, false, bvf, (short)0, oa[nt], false, false);
    }
  }
#pragma unroll
  for (int nt = 0; nt < 4; ++nt)
#pragma unroll
    for (int r = 0; r < 8; ++r)
      O[(qrow0 + r + 8 * half) * E_ + h * HD_ + nt * 16 + nn] = (bf16_t)oa[nt][r];
}

// ====== out-proj GEMM fused with AdaLN modulation + residual + transpose ====
__global__ __launch_bounds__(256) void outproj_kernel(
    const bf16_t* __restrict__ O, const float* __restrict__ Wo,
    const float* __restrict__ bo, const float* __restrict__ mod,
    const float* __restrict__ img, float* __restrict__ out) {
  __shared__ __align__(32) bf16_t As[128 * 32];
  __shared__ __align__(32) bf16_t Ws[32 * 64];   // frag-major
  const int t = threadIdx.x, lane = t & 31, w = t >> 5;
  const int half = lane >> 4, nn = lane & 15;
  const int m0 = blockIdx.x * 128, n0 = blockIdx.y * 64;
  const int cW = t & 63;
  const int oct = t >> 6;
  const int wdst = (((cW >> 4) * 32) + ((oct >> 1) * 16) + (cW & 15)) * 16 + (oct & 1) * 8;
  v8f acc[4] = {};
  for (int kk = 0; kk < 256; kk += 32) {
    {
      int row = t >> 1, seg = (t & 1) * 16;
      const bf16_t* src = O + (size_t)(m0 + row) * 256 + kk + seg;
      st8(As + row * 32 + seg, ld8(src));
      st8(As + row * 32 + seg + 8, ld8(src + 8));
      if (kk + 32 < 256) __builtin_prefetch(src + 32, 0, 1);
    }
    {
      const float* wsrc = Wo + (size_t)(kk + oct * 8) * 256 + n0 + cW;
      v8bf tmp;
#pragma unroll
      for (int q = 0; q < 8; ++q) tmp[q] = (bf16_t)wsrc[(size_t)q * 256];
      st8(Ws + wdst, tmp);
    }
    __syncthreads();
    v16bf a = load_afrag(As + (w * 16 + nn) * 32, half);
#pragma unroll
    for (int nt = 0; nt < 4; ++nt) {
      v16bf bfr = load_bfrag(Ws + (nt * 32 + lane) * 16);
      acc[nt] = __builtin_amdgcn_wmma_f32_16x16x32_bf16(
          false, a, false, bfr, (short)0, acc[nt], false, false);
    }
    __syncthreads();
  }
#pragma unroll
  for (int nt = 0; nt < 4; ++nt)
#pragma unroll
    for (int r = 0; r < 8; ++r) {
      int row = m0 + w * 16 + r + 8 * half;       // global token index
      int b = row >> 12, s = row & 4095;
      int c = n0 + nt * 16 + nn;
      float a0 = acc[nt][r] + bo[c];
      float shift = mod[b * 768 + c];
      float scl   = mod[b * 768 + 256 + c];
      float gate  = mod[b * 768 + 512 + c];
      float xres  = img[((size_t)b * C_ + c) * S_ + s];
      out[((size_t)b * C_ + c) * S_ + s] = gate * (a0 * (1.f + scl) + shift) + xres;
    }
}

// ============================== launcher ====================================
extern "C" void kernel_launch(void* const* d_in, const int* in_sizes, int n_in,
                              void* d_out, int out_size, void* d_ws, size_t ws_size,
                              hipStream_t stream) {
  const float* img  = (const float*)d_in[0];
  const float* refs = (const float*)d_in[1];
  const unsigned char* masks = (const unsigned char*)d_in[2];
  const float* re  = (const float*)d_in[3];
  const float* liw = (const float*)d_in[4];
  const float* lib = (const float*)d_in[5];
  const float* ltw = (const float*)d_in[6];
  const float* ltb = (const float*)d_in[7];
  const float* Wq = (const float*)d_in[8];  const float* bq  = (const float*)d_in[9];
  const float* Wk = (const float*)d_in[10]; const float* bk  = (const float*)d_in[11];
  const float* Wv = (const float*)d_in[12]; const float* bvv = (const float*)d_in[13];
  const float* Wo = (const float*)d_in[14]; const float* bo  = (const float*)d_in[15];
  const float* Wa = (const float*)d_in[16]; const float* ba  = (const float*)d_in[17];
  float* out = (float*)d_out;

  char* ws = (char*)d_ws;
  bf16_t* qn = (bf16_t*)(ws);                       // 32 MB (reused as O)
  bf16_t* Qb = (bf16_t*)(ws + (size_t)33554432);    // 32 MB
  bf16_t* kn = (bf16_t*)(ws + (size_t)67108864);    // 2 MB
  bf16_t* Kb = (bf16_t*)(ws + (size_t)69206016);    // 2 MB
  bf16_t* Vb = (bf16_t*)(ws + (size_t)71303168);    // 2 MB
  float* mod = (float*)(ws + (size_t)73400320);     // 48 KB
  bf16_t* Ob = qn;  // qn is dead after Q projection

  const size_t attn_lds = (size_t)(64 * VSTRIDE + 8 * 16 * 256) * sizeof(bf16_t);

  ln_img_kernel<<<dim3(16, 64), 256, 256 * 65 * sizeof(float), stream>>>(img, liw, lib, qn);
  ln_txt_kernel<<<dim3(512), 256, 0, stream>>>(refs, ltw, ltb, kn);
  mod_kernel<<<dim3(16, 3), 256, 0, stream>>>(re, Wa, ba, mod);
  gemm_bf16_256<<<dim3(512, 4), 256, 0, stream>>>(qn, Wq, bq, Qb);
  gemm_bf16_256<<<dim3(32, 4), 256, 0, stream>>>(kn, Wk, bk, Kb);
  gemm_bf16_256<<<dim3(32, 4), 256, 0, stream>>>(kn, Wv, bvv, Vb);
  attn_kernel<<<dim3(16, 4, 32), 256, attn_lds, stream>>>(Qb, Kb, Vb, masks, Ob);
  outproj_kernel<<<dim3(512, 4), 256, 0, stream>>>(Ob, Wo, bo, mod, img, out);
}
